// SimpleAbsorbedAttention_78280073937161
// MI455X (gfx1250) — matrix operand order repaired and checked by reading, not compiled, not run on hardware
//
#include <hip/hip_runtime.h>

#define H_   128
#define K_   128
#define R_   512
#define DQ_  1536
#define B_   32
#define L_   4096
#define HK_  (H_ * K_)

typedef __attribute__((ext_vector_type(16))) __bf16 v16bf;
typedef __attribute__((ext_vector_type(8)))  float  v8f;
typedef __attribute__((ext_vector_type(2)))  __bf16 v2bf;

#define WMMA_BF16(a, b, c) \
  __builtin_amdgcn_wmma_f32_16x16x32_bf16(false, (a), false, (b), (short)0, (c), false, false)

#if defined(__has_builtin)
#if __has_builtin(__builtin_amdgcn_cvt_pk_bf16_f32)
#define HAVE_CVT_PK 1
#endif
#endif

union BF16Frag {
  v16bf v;
  unsigned short s[16];
  unsigned d[8];
};

// pack two f32 -> one dword holding {bf16(f1) : bf16(f0)} (f0 in low half)
__device__ __forceinline__ unsigned pack2_bf16(float f0, float f1) {
#ifdef HAVE_CVT_PK
  v2bf t = __builtin_amdgcn_cvt_pk_bf16_f32(f0, f1);
  unsigned r;
  __builtin_memcpy(&r, &t, 4);
  return r;
#else
  // round-half-up then take high 16 bits of each; single v_perm_b32 packs both
  unsigned a = __float_as_uint(f0) + 0x8000u;
  unsigned b = __float_as_uint(f1) + 0x8000u;
  return __builtin_amdgcn_perm(b, a, 0x07060302u);
#endif
}

__device__ __forceinline__ unsigned short f2bf1(float f) {
  return (unsigned short)((__float_as_uint(f) + 0x8000u) >> 16);
}

__device__ __forceinline__ v16bf pack16(const float* t) {
  BF16Frag f;
#pragma unroll
  for (int i = 0; i < 8; ++i) f.d[i] = pack2_bf16(t[2 * i], t[2 * i + 1]);
  return f.v;
}

// A-fragment (16x32 bf16): lane holds k = base + {0..15} (kk=(lane>>4)*16
// folded into the pointer by the caller).
__device__ __forceinline__ v16bf load_fragA_f32(const float* __restrict__ p) {
  float t[16];
#pragma unroll
  for (int e = 0; e < 16; ++e) t[e] = p[e];
  return pack16(t);
}

// B-fragment (32x16 bf16): lane holds 16 contiguous k (kk2=(lane>>4)*16 folded in).
__device__ __forceinline__ v16bf load_fragB_f32(const float* __restrict__ p) {
  float t[16];
#pragma unroll
  for (int e = 0; e < 16; ++e) t[e] = p[e];
  return pack16(t);
}

__device__ __forceinline__ v16bf load_fragB_f32_strided(const float* __restrict__ p,
                                                        int stride) {
  float t[16];
#pragma unroll
  for (int e = 0; e < 16; ++e) t[e] = p[(size_t)e * stride];
  return pack16(t);
}

// ---------------- Kernel A: q = hidden @ Wq^T + bq  (M=32, N=16384, K=1536) ----
__global__ void __launch_bounds__(32) qproj_kernel(
    const float* __restrict__ hid, const float* __restrict__ Wq,
    const float* __restrict__ bq, float* __restrict__ q) {
  const int lane = threadIdx.x;
  const int nn = lane & 15, kkA = (lane >> 4) * 16, kkB = (lane >> 4) * 16;
  const int mb = (lane >> 4) * 8;
  const int n0 = blockIdx.x * 16;
  const int m0 = blockIdx.y * 16;
  const float bias = bq[n0 + nn];
  v8f acc;
#pragma unroll
  for (int j = 0; j < 8; ++j) acc[j] = bias;
  const float* arow = hid + (size_t)(m0 + nn) * DQ_ + kkA;
  const float* bcol = Wq + (size_t)(n0 + nn) * DQ_ + kkB;  // B[k][n] = Wq[n][k]
#pragma unroll 4
  for (int kb = 0; kb < DQ_; kb += 32) {
    v16bf a  = load_fragA_f32(arow + kb);
    v16bf bm = load_fragB_f32(bcol + kb);
    acc = WMMA_BF16(a, bm, acc);
  }
#pragma unroll
  for (int j = 0; j < 8; ++j)
    q[(size_t)(m0 + mb + j) * HK_ + n0 + nn] = acc[j];
}

// ---------------- Kernel B: q_abs[b,h,r] = sum_k w_kc[h,k,r] * q[b,h,k] -------
__global__ void __launch_bounds__(32) qabs_kernel(
    const float* __restrict__ q, const float* __restrict__ w_kc,
    float* __restrict__ qabs) {
  const int lane = threadIdx.x;
  const int nn = lane & 15, kkA = (lane >> 4) * 16, kkB = (lane >> 4) * 16;
  const int mb = (lane >> 4) * 8;
  const int h  = blockIdx.x;
  const int r0 = blockIdx.y * 16;
  const int m0 = blockIdx.z * 16;
  v8f acc;
#pragma unroll
  for (int j = 0; j < 8; ++j) acc[j] = 0.f;
  const float* arow  = q + (size_t)(m0 + nn) * HK_ + h * K_ + kkA;
  const float* bbase = w_kc + ((size_t)h * K_ + kkB) * R_ + r0 + nn;
#pragma unroll
  for (int kb = 0; kb < K_; kb += 32) {
    v16bf a  = load_fragA_f32(arow + kb);
    v16bf bm = load_fragB_f32_strided(bbase + (size_t)kb * R_, R_);
    acc = WMMA_BF16(a, bm, acc);
  }
#pragma unroll
  for (int j = 0; j < 8; ++j)
    qabs[((size_t)(m0 + mb + j) * H_ + h) * R_ + r0 + nn] = acc[j];
}

// ---------------- Kernel C: fused streaming softmax-attention over L ----------
#define LC        64    // L-chunk rows
#define WS_BF     520   // bf16 elems per working row (512 + 8 pad -> 4-bank skew)
#define P_STRIDE  72    // bf16 elems per LDS P row (64 + 8 pad)

__global__ void __launch_bounds__(128) attn_kernel(
    const float* __restrict__ qabs, const float* __restrict__ kv,
    float* __restrict__ ctx) {
  extern __shared__ char smem[];
  float* stage = (float*)smem;                                  // 64*512*4 = 128 KB
  unsigned short* work = (unsigned short*)(smem + (size_t)LC * R_ * 4);  // 64*520*2
  unsigned short* pLds =
      (unsigned short*)(smem + (size_t)LC * R_ * 4 + (size_t)LC * WS_BF * 2);
  float* redLds = (float*)(smem + (size_t)LC * R_ * 4 + (size_t)LC * WS_BF * 2 +
                           16 * P_STRIDE * 2);

  const int tid  = threadIdx.x;
  const int w    = tid >> 5;
  const int lane = tid & 31;
  const int nn   = lane & 15;
  const int kkA  = (lane >> 4) * 16;
  const int kkB  = (lane >> 4) * 16;
  const int mb   = (lane >> 4) * 8;
  const int h0   = blockIdx.x * 16;
  const int b    = blockIdx.y;

  // Async DMA of one KV chunk [LC x 512] f32 -> unpadded LDS staging buffer.
  // Row stride is 2048 B on BOTH sides, so a single shared IOFFSET immediate
  // addresses both (per ISA: same INST_OFFSET added to LDS and global address).
  const unsigned stage_base = (unsigned)(uintptr_t)stage + tid * 16;
  auto issue_chunk = [&](const float* src) {
    const float* gp = src + tid * 4;                 // 4 f32 = 16B per lane
#pragma unroll
    for (int i = 0; i < LC; ++i) {                   // one row per instruction
      asm volatile("global_load_async_to_lds_b128 %0, %1, off offset:%2"
                   :: "v"(stage_base), "v"(gp), "i"(i * (R_ * 4)) : "memory");
    }
  };

  // Q A-fragments: [16 heads x R=512] bf16, 128 VGPRs, held for the whole kernel
  v16bf qa[16];
  {
    const float* qrow = qabs + ((size_t)b * H_ + h0 + nn) * R_ + kkA;
#pragma unroll
    for (int f = 0; f < 16; ++f) qa[f] = load_fragA_f32(qrow + f * 32);
  }

  // ctx accumulator: wave w owns r in [w*128, w*128+128): 8 x 16x16 f32 tiles
  v8f acc[8];
#pragma unroll
  for (int t = 0; t < 8; ++t)
#pragma unroll
    for (int j = 0; j < 8; ++j) acc[t][j] = 0.f;

  float rmax[8], rsum[8];
#pragma unroll
  for (int j = 0; j < 8; ++j) { rmax[j] = -3.0e38f; rsum[j] = 0.f; }

  const float* kvb = kv + (size_t)b * L_ * R_;

  issue_chunk(kvb);                                   // prefetch chunk 0

  for (int c = 0; c < L_ / LC; ++c) {
    asm volatile("s_wait_asynccnt 0" ::: "memory");   // my chunk-c DMAs done
    __syncthreads();                                  // everyone's DMAs visible

    // ---- convert chunk f32 -> bf16 exactly once (stage -> work) ----
#pragma unroll 4
    for (int i = 0; i < (LC * R_ / 8) / 128; ++i) {   // 32 iters, 8 f32 each
      int idx = i * 128 + tid;
      int row = idx >> 6;                             // 64 groups-of-8 per row
      int c8  = (idx & 63) * 8;
      const float* sp = stage + row * R_ + c8;
      float4 a = *(const float4*)sp;
      float4 d = *(const float4*)(sp + 4);
      uint4 o;
      o.x = pack2_bf16(a.x, a.y);
      o.y = pack2_bf16(a.z, a.w);
      o.z = pack2_bf16(d.x, d.y);
      o.w = pack2_bf16(d.z, d.w);
      *(uint4*)&work[row * WS_BF + c8] = o;
    }
    __syncthreads();                                  // work ready, stage free

    if (c + 1 < L_ / LC)                              // overlap next chunk's DMA
      issue_chunk(kvb + (size_t)(c + 1) * LC * R_);

    // ---- S tile = Q x KV^T : wave w covers columns l = c*LC + w*16 + nn ----
    v8f s;
#pragma unroll
    for (int j = 0; j < 8; ++j) s[j] = 0.f;
    {
      const unsigned short* prow = work + (w * 16 + nn) * WS_BF;
#pragma unroll
      for (int f = 0; f < 16; ++f) {                  // R = 512 = 16 * 32
        const unsigned short* p = prow + f * 32 + kkB;
        BF16Frag bf;
        *(uint4*)&bf.s[0] = *(const uint4*)p;         // k = +0..7
        *(uint4*)&bf.s[8] = *(const uint4*)(p + 8);   // k = +8..15
        s = WMMA_BF16(qa[f], bf.v, s);
      }
    }

    // wave-local row max over this wave's 16 columns (reduce over n-lanes)
    float cmax[8];
#pragma unroll
    for (int j = 0; j < 8; ++j) cmax[j] = s[j];
#pragma unroll
    for (int d = 1; d < 16; d <<= 1)
#pragma unroll
      for (int j = 0; j < 8; ++j) cmax[j] = fmaxf(cmax[j], __shfl_xor(cmax[j], d, 32));
    if (nn == 0) {
#pragma unroll
      for (int j = 0; j < 8; ++j) redLds[w * 16 + mb + j] = cmax[j];
    }
    __syncthreads();

    // chunk-global max; identical online-rescale computed by every wave
    float scale[8];
#pragma unroll
    for (int j = 0; j < 8; ++j) {
      float cm = fmaxf(fmaxf(redLds[mb + j], redLds[16 + mb + j]),
                       fmaxf(redLds[32 + mb + j], redLds[48 + mb + j]));
      float nm = fmaxf(rmax[j], cm);
      scale[j] = __expf(rmax[j] - nm);
      rmax[j] = nm;
      rsum[j] *= scale[j];
    }
#pragma unroll
    for (int t = 0; t < 8; ++t)
#pragma unroll
      for (int j = 0; j < 8; ++j) acc[t][j] *= scale[j];

    // P = exp(S - max): bf16 into LDS; accumulate this wave's row partial sums
    float ps[8];
#pragma unroll
    for (int j = 0; j < 8; ++j) {
      float p = __expf(s[j] - rmax[j]);
      ps[j] = p;
      pLds[(mb + j) * P_STRIDE + w * 16 + nn] = f2bf1(p);
    }
#pragma unroll
    for (int d = 1; d < 16; d <<= 1)
#pragma unroll
      for (int j = 0; j < 8; ++j) ps[j] += __shfl_xor(ps[j], d, 32);
#pragma unroll
    for (int j = 0; j < 8; ++j) rsum[j] += ps[j];
    __syncthreads();

    // ---- ctx += P[16xLC] x KV[LC x r-slice] (wave w: r in [w*128, +128)) ----
#pragma unroll
    for (int kc = 0; kc < 2; ++kc) {          // LC = 64 = 2 * 32
      BF16Frag af;                            // P as A-matrix: m = head, k = l
      const unsigned short* pa = &pLds[nn * P_STRIDE + kc * 32 + kkA];
      *(uint4*)&af.s[0] = *(const uint4*)pa;
      *(uint4*)&af.s[8] = *(const uint4*)(pa + 8);
#pragma unroll
      for (int t = 0; t < 8; ++t) {
        const int r = w * 128 + t * 16 + nn;
        const unsigned short* pb = work + (kc * 32 + kkB) * WS_BF + r;
        BF16Frag bf;                          // KV as B-matrix: k = l, n = r
#pragma unroll
        for (int e = 0; e < 16; ++e) bf.s[e] = pb[e * WS_BF];
        acc[t] = WMMA_BF16(af.v, bf.v, acc[t]);
      }
    }
    __syncthreads();
  }

  // total row sums across the 4 waves, then normalize and write ctx
  if (nn == 0) {
#pragma unroll
    for (int j = 0; j < 8; ++j) redLds[w * 16 + mb + j] = rsum[j];
  }
  __syncthreads();
  float tot[8];
#pragma unroll
  for (int j = 0; j < 8; ++j)
    tot[j] = (redLds[mb + j] + redLds[16 + mb + j]) +
             (redLds[32 + mb + j] + redLds[48 + mb + j]);

#pragma unroll
  for (int t = 0; t < 8; ++t) {
    const int r = w * 128 + t * 16 + nn;
#pragma unroll
    for (int j = 0; j < 8; ++j)
      ctx[((size_t)b * H_ + h0 + mb + j) * R_ + r] = acc[t][j] / tot[j];
  }
}

// ---------------- Kernel D: out[b,h,k] = sum_r w_vc[h,k,r] * ctx[b,h,r] ------
__global__ void __launch_bounds__(32) outproj_kernel(
    const float* __restrict__ ctx, const float* __restrict__ w_vc,
    float* __restrict__ out) {
  const int lane = threadIdx.x;
  const int nn = lane & 15, kkA = (lane >> 4) * 16, kkB = (lane >> 4) * 16;
  const int mb = (lane >> 4) * 8;
  const int h  = blockIdx.x;
  const int n0 = blockIdx.y * 16;   // k_out tile
  const int m0 = blockIdx.z * 16;   // batch tile
  v8f acc;
#pragma unroll
  for (int j = 0; j < 8; ++j) acc[j] = 0.f;
  const float* arow = ctx + ((size_t)(m0 + nn) * H_ + h) * R_ + kkA;
  const float* bcol = w_vc + ((size_t)h * K_ + n0 + nn) * R_ + kkB;
#pragma unroll 4
  for (int kb = 0; kb < R_; kb += 32) {
    v16bf a  = load_fragA_f32(arow + kb);
    v16bf bm = load_fragB_f32(bcol + kb);
    acc = WMMA_BF16(a, bm, acc);
  }
#pragma unroll
  for (int j = 0; j < 8; ++j)
    out[(size_t)(m0 + mb + j) * HK_ + h * K_ + n0 + nn] = acc[j];
}

extern "C" void kernel_launch(void* const* d_in, const int* in_sizes, int n_in,
                              void* d_out, int out_size, void* d_ws, size_t ws_size,
                              hipStream_t stream) {
  (void)in_sizes; (void)n_in; (void)out_size; (void)ws_size;
  const float* hid  = (const float*)d_in[0];
  const float* kv   = (const float*)d_in[1];
  const float* Wq   = (const float*)d_in[2];
  const float* bq   = (const float*)d_in[3];
  const float* w_kc = (const float*)d_in[4];
  const float* w_vc = (const float*)d_in[5];
  float* out = (float*)d_out;

  char* ws = (char*)d_ws;
  float* q_ws    = (float*)ws;                                             // 2 MB
  float* qabs_ws = (float*)(ws + (size_t)B_ * HK_ * sizeof(float));        // 8 MB
  float* ctx_ws  = (float*)(ws + (size_t)B_ * HK_ * sizeof(float)
                               + (size_t)B_ * H_ * R_ * sizeof(float));    // 8 MB

  qproj_kernel<<<dim3(HK_ / 16, B_ / 16), 32, 0, stream>>>(hid, Wq, bq, q_ws);
  qabs_kernel<<<dim3(H_, R_ / 16, B_ / 16), 32, 0, stream>>>(q_ws, w_kc, qabs_ws);
  size_t smem = (size_t)LC * R_ * 4 + (size_t)LC * WS_BF * 2 +
                16 * P_STRIDE * 2 + 64 * sizeof(float);
  attn_kernel<<<dim3(H_ / 16, B_), 128, smem, stream>>>(qabs_ws, kv, ctx_ws);
  outproj_kernel<<<dim3(H_, K_ / 16, B_ / 16), 32, 0, stream>>>(ctx_ws, w_vc, out);
}
